// PNP_51204600103534
// MI455X (gfx1250) — matrix operand order repaired and checked by reading, not compiled, not run on hardware
//
#include <hip/hip_runtime.h>
#include <math.h>

// ---------------------------------------------------------------------------
// Types for CDNA5 WMMA (wave32, 16x16x32 bf16 -> f32)
// ---------------------------------------------------------------------------
typedef __bf16 bf16;
typedef bf16  v16bf __attribute__((ext_vector_type(16)));
typedef float v8f   __attribute__((ext_vector_type(8)));

// Explicit global (addrspace 1) pointers: keeps weight streams on the
// global_load path instead of flat_load (flat ticks DScnt + LDS arbitration).
typedef const v16bf __attribute__((address_space(1)))* gv16p;
typedef const float __attribute__((address_space(1)))* gfloatp;
typedef float       __attribute__((address_space(1)))* gfloatp_mut;

#define T_STEPS 512
#define HDIM    512
#define NWAVES  32
#define NTHR    1024

__device__ __constant__ int c_par[24] = {-1,0,0,0,1,2,3,4,5,6,7,8,9,9,9,12,13,14,16,17,18,19,20,21};
__device__ __constant__ int c_red[15] = {1,2,3,4,5,6,9,12,13,14,15,16,17,18,19};
__device__ __constant__ int c_ign[9]  = {0,7,8,10,11,20,21,22,23};

__device__ inline float sigf(float x) { return 1.0f / (1.0f + __expf(-x)); }

// ---------------------------------------------------------------------------
// Workspace layout: compile-time function of fixed model dims, shared by host
// packing code and device kernels via constant memory.
// ---------------------------------------------------------------------------
struct WsLayout {
  unsigned long long Wl[15], bl[15];   // LSTM layer packed W / combined bias
  unsigned long long Wh[5],  bh[5];    // head packed W / padded bias
  unsigned long long Wm[4],  bm[4];    // mlpA packed W / padded bias
  unsigned long long cst, total;
};

constexpr unsigned long long ws_al(unsigned long long x) { return (x + 255ull) & ~255ull; }

constexpr WsLayout make_layout() {
  WsLayout L{};
  const int in_pad[5]  = {64, 64, 128, 160, 160};
  const int head_mt[5] = {1, 5, 6, 5, 1};
  const int mlp_mt[4]  = {32, 32, 32, 1};
  const int mlp_kt[4]  = {4, 16, 16, 16};
  unsigned long long p = 0;
  for (int s = 0; s < 5; ++s) {
    for (int l = 0; l < 3; ++l) {
      int Kp = ((l == 0) ? in_pad[s] : HDIM) + HDIM;
      L.Wl[s * 3 + l] = p; p = ws_al(p + 2048ull * Kp * 2);
      L.bl[s * 3 + l] = p; p = ws_al(p + 2048ull * 4);
    }
    L.Wh[s] = p; p = ws_al(p + (unsigned long long)head_mt[s] * 16 * 512 * 2);
    L.bh[s] = p; p = ws_al(p + (unsigned long long)head_mt[s] * 16 * 4);
  }
  for (int l = 0; l < 4; ++l) {
    L.Wm[l] = p; p = ws_al(p + (unsigned long long)mlp_mt[l] * mlp_kt[l] * 512 * 2);
    L.bm[l] = p; p = ws_al(p + (unsigned long long)mlp_mt[l] * 16 * 4);
  }
  L.cst = p; p = ws_al(p + 15ull * HDIM * 4);
  L.total = p;
  return L;
}

__device__ __constant__ WsLayout d_lay = make_layout();

// ---------------------------------------------------------------------------
// WMMA matvec: y[Mt*16] = Wpacked(Mt*16 x Kt*32, bf16 fragment-major) @ x + bias
// x replicated across all 16 B columns -> every D column holds the matvec;
// lanes 0 / 16 commit rows 0-7 / 8-15 of each 16-row tile.
// Weight/bias addresses passed as integers and cast to addrspace(1) so the
// backend emits global_load_b128 (not flat_load) across the noinline boundary.
// ---------------------------------------------------------------------------
__device__ inline void matvec(unsigned long long Wa, unsigned long long Ba,
                              int Mt, int Kt, const bf16* xbf, float* y, int tid) {
  const int lane = tid & 31, wv = tid >> 5, hi = lane >> 4;
  const bf16* xb = xbf + hi * 16;
  gv16p   Wg   = (gv16p)Wa;
  gfloatp bias = (gfloatp)Ba;
  for (int mt = wv; mt < Mt; mt += NWAVES) {
    v8f acc = {};
    gv16p wfrag = Wg + (size_t)mt * Kt * 32 + lane;
    for (int kt = 0; kt < Kt; ++kt) {
      v16bf afrag = wfrag[(size_t)kt * 32];
      v16bf bfrag = *reinterpret_cast<const v16bf*>(xb + kt * 32);
      acc = __builtin_amdgcn_wmma_f32_16x16x32_bf16(false, afrag, false, bfrag,
                                                    (short)0, acc, false, false);
    }
    if ((lane & 15) == 0) {
      int mb = mt * 16 + hi * 8;
#pragma unroll
      for (int i = 0; i < 8; ++i) y[mb + i] = acc[i] + bias[mb + i];
    }
  }
}

// ---------------------------------------------------------------------------
// Packing kernels (run each launch; deterministic)
// ---------------------------------------------------------------------------
__global__ void pack_lstm_k(const float* __restrict__ Wih, const float* __restrict__ Whh,
                            int in_act, int in_pad, bf16* __restrict__ dst) {
  const int Kp = in_pad + HDIM, Kt = Kp / 32;
  const size_t total = (size_t)128 * Kt * 512;
  for (size_t idx = (size_t)blockIdx.x * blockDim.x + threadIdx.x; idx < total;
       idx += (size_t)gridDim.x * blockDim.x) {
    int j = (int)(idx & 15);
    int lane = (int)((idx >> 4) & 31);
    size_t frag = idx >> 9;
    int kt = (int)(frag % Kt), mt = (int)(frag / Kt);
    int m = mt * 16 + (lane & 15);
    int hi = lane >> 4;
    int koff = hi * 16 + j;
    int K = kt * 32 + koff;
    float v;
    if (K < in_pad) v = (K < in_act) ? Wih[(size_t)m * in_act + K] : 0.0f;
    else            v = Whh[(size_t)m * HDIM + (K - in_pad)];
    dst[idx] = (bf16)v;
  }
}

__global__ void pack_mat_k(const float* __restrict__ W, int fo, int fi,
                           int Mt, int Kt, bf16* __restrict__ dst) {
  const size_t total = (size_t)Mt * Kt * 512;
  for (size_t idx = (size_t)blockIdx.x * blockDim.x + threadIdx.x; idx < total;
       idx += (size_t)gridDim.x * blockDim.x) {
    int j = (int)(idx & 15);
    int lane = (int)((idx >> 4) & 31);
    size_t frag = idx >> 9;
    int kt = (int)(frag % Kt), mt = (int)(frag / Kt);
    int m = mt * 16 + (lane & 15);
    int hi = lane >> 4;
    int koff = hi * 16 + j;
    int K = kt * 32 + koff;
    float v = (m < fo && K < fi) ? W[(size_t)m * fi + K] : 0.0f;
    dst[idx] = (bf16)v;
  }
}

__global__ void bias_comb_k(const float* a, const float* b, float* dst) {
  int i = blockIdx.x * blockDim.x + threadIdx.x;
  if (i < 4 * HDIM) dst[i] = a[i] + b[i];
}
__global__ void bias_pad_k(const float* b, int n, int np, float* dst) {
  int i = blockIdx.x * blockDim.x + threadIdx.x;
  if (i < np) dst[i] = (i < n) ? b[i] : 0.0f;
}
__global__ void cinit_k(const float* c0, float* cst) {
  int i = blockIdx.x * blockDim.x + threadIdx.x;
  if (i < 15 * HDIM) cst[i] = (i < 3 * HDIM) ? c0[i] : 0.0f;
}

// ---------------------------------------------------------------------------
// Persistent sequential scan kernel (single WGP, 32 waves, LDS-resident state)
// ---------------------------------------------------------------------------
struct SeqArgs {
  char* ws;                       // workspace base; offsets come from d_lay
  const float *a, *w, *R, *t_local, *pRB0, *pl1_h0;
  float* out;
};

struct SeqShared {
  float h[15][HDIM];
  float g[4 * HDIM];
  float yb[HDIM];
  float xv[160];
  float aRB[18], wRB[18], RRB[54];
  float corr[16], pRBc[16], prevp[16], pRBn[16], prevw[4];
  float rbuf[96], red[15 * 9];
  float glb[24 * 9], pose[24 * 9], Rg[24 * 9], pj[24 * 3], aj[72];
  float tl[72], avb[72], cfg2[2], RIR[9];
};

// noinline: one code copy instead of 5 inlined expansions -> short pointer
// live-ranges, far less spill pressure in the persistent loop.
__device__ __attribute__((noinline)) void lstm3(unsigned long long wsa, SeqShared& S, bf16* xbf,
                                                int sid, const float* x0, int in_act, int in_pad,
                                                int tid) {
  for (int l = 0; l < 3; ++l) {
    const int slot = sid * 3 + l;
    const int ia = (l == 0) ? in_act : HDIM;
    const int ip = (l == 0) ? in_pad : HDIM;
    const float* xin = (l == 0) ? x0 : S.h[slot - 1];
    const int Kp = ip + HDIM;
    for (int k = tid; k < Kp; k += NTHR)
      xbf[k] = (bf16)((k < ip) ? ((k < ia) ? xin[k] : 0.0f) : S.h[slot][k - ip]);
    __syncthreads();
    matvec(wsa + d_lay.Wl[slot], wsa + d_lay.bl[slot], 128, Kp / 32, xbf, S.g, tid);
    __syncthreads();
    if (tid < HDIM) {
      float gi = S.g[tid], gf = S.g[HDIM + tid], gg = S.g[2 * HDIM + tid], go = S.g[3 * HDIM + tid];
      gfloatp_mut cp = (gfloatp_mut)(wsa + d_lay.cst) + slot * HDIM + tid;
      float cn = sigf(gf) * cp[0] + sigf(gi) * tanhf(gg);
      cp[0] = cn;
      S.h[slot][tid] = sigf(go) * tanhf(cn);
    }
    __syncthreads();
  }
}

__device__ __attribute__((noinline)) void headmv(unsigned long long wsa, SeqShared& S, bf16* xbf,
                                                 int sid, int Mt, int tid) {
  if (tid < HDIM) xbf[tid] = (bf16)S.h[sid * 3 + 2][tid];
  __syncthreads();
  matvec(wsa + d_lay.Wh[sid], wsa + d_lay.bh[sid], Mt, 16, xbf, S.yb, tid);
  __syncthreads();
}

__launch_bounds__(NTHR, 1)
__global__ void seq_kernel(SeqArgs A) {
  __shared__ SeqShared S;
  __shared__ bf16 xbf[1088] __attribute__((aligned(32)));
  const int tid = threadIdx.x;
  const unsigned long long wsa = (unsigned long long)A.ws;
  const float invdt = 60.0f;

  // ---- carry init ----
  for (int i = tid; i < 15 * HDIM; i += NTHR)
    S.h[i / HDIM][i % HDIM] = (i < 3 * HDIM) ? A.pl1_h0[i] : 0.0f;
  if (tid < 72) S.tl[tid] = A.t_local[tid];
  if (tid < 15) { S.prevp[tid] = 0.0f; S.pRBc[tid] = A.pRB0[tid]; }
  if (tid < 3)  S.prevw[tid] = 0.0f;
  __syncthreads();

  for (int t = 0; t < T_STEPS; ++t) {
    const float* Rt = A.R + (size_t)t * 54;
    const float* at = A.a + (size_t)t * 18;
    const float* wt = A.w + (size_t)t * 18;
    if (tid < 9) S.RIR[tid] = Rt[45 + tid];
    __syncthreads();
    if (tid < 18) {
      int k = tid / 3, i = tid % 3;
      float s1 = 0, s2 = 0;
      for (int j = 0; j < 3; ++j) { s1 += S.RIR[j * 3 + i] * at[k * 3 + j]; s2 += S.RIR[j * 3 + i] * wt[k * 3 + j]; }
      S.aRB[tid] = s1; S.wRB[tid] = s2;
    }
    if (tid < 54) {
      int k = tid / 9, r = (tid % 9) / 3, m = tid % 3;
      float s = 0;
      for (int j = 0; j < 3; ++j) s += S.RIR[j * 3 + r] * Rt[k * 9 + j * 3 + m];
      S.RRB[tid] = s;
    }
    __syncthreads();

    // ---- x108 ----
    if (tid < 108) {
      float v;
      if (tid < 18)      v = S.aRB[tid] * 0.05f;
      else if (tid < 72) v = S.RRB[tid - 18];
      else if (tid < 75) v = S.wRB[15 + tid - 72] * 0.25f;
      else if (tid < 78) { int i = tid - 75; v = (t > 0) ? (S.wRB[15 + i] - S.prevw[i]) * (invdt / 400.0f) : 0.0f; }
      else if (tid < 93) { int i = tid - 78; v = (t > 0) ? (S.pRBc[i] - S.prevp[i]) * (invdt * 0.5f) : 0.0f; }
      else               v = S.pRBc[tid - 93];
      S.xv[tid] = v;
    }
    __syncthreads();

    // ---- mlpA ----
    if (tid < 128) xbf[tid] = (bf16)((tid < 108) ? S.xv[tid] : 0.0f);
    __syncthreads();
    matvec(wsa + d_lay.Wm[0], wsa + d_lay.bm[0], 32, 4, xbf, S.yb, tid);
    __syncthreads();
    for (int lyr = 1; lyr <= 2; ++lyr) {
      if (tid < HDIM) xbf[tid] = (bf16)fmaxf(S.yb[tid], 0.0f);
      __syncthreads();
      matvec(wsa + d_lay.Wm[lyr], wsa + d_lay.bm[lyr], 32, 16, xbf, S.yb, tid);
      __syncthreads();
    }
    if (tid < HDIM) xbf[tid] = (bf16)fmaxf(S.yb[tid], 0.0f);
    __syncthreads();
    matvec(wsa + d_lay.Wm[3], wsa + d_lay.bm[3], 1, 16, xbf, S.yb, tid);
    __syncthreads();
    if (tid < 15) S.corr[tid] = S.yb[tid] * 5.0f;
    __syncthreads();

    // ---- imu_dyn (60) -> pl1 -> pRB_new ----
    if (tid < 60) {
      float v;
      if (tid < 15) { int k = tid / 3, i = tid % 3; v = (S.aRB[k * 3 + i] - S.aRB[15 + i] - S.corr[tid]) * 0.05f; }
      else v = S.RRB[tid - 15];
      S.xv[tid] = v;
    }
    __syncthreads();
    lstm3(wsa, S, xbf, 0, S.xv, 60, 64, tid);
    headmv(wsa, S, xbf, 0, 1, tid);
    if (tid < 15) S.pRBn[tid] = S.yb[tid];
    __syncthreads();

    // ---- x60 -> ik1 -> jpos ----
    if (tid < 60) S.xv[tid] = (tid < 45) ? S.RRB[tid] : S.pRBn[tid - 45];
    __syncthreads();
    lstm3(wsa, S, xbf, 1, S.xv, 60, 64, tid);
    headmv(wsa, S, xbf, 1, 5, tid);

    // ---- x114 -> ik2 -> r6 ----
    if (tid < 114) S.xv[tid] = (tid < 45) ? S.RRB[tid] : S.yb[tid - 45];
    __syncthreads();
    lstm3(wsa, S, xbf, 2, S.xv, 114, 128, tid);
    headmv(wsa, S, xbf, 2, 6, tid);
    if (tid < 90) S.rbuf[tid] = S.yb[tid];
    __syncthreads();

    // ---- r6d_to_rm (15 rotations) ----
    if (tid < 15) {
      const float* r = S.rbuf + tid * 6;
      float n1 = rsqrtf(r[0] * r[0] + r[1] * r[1] + r[2] * r[2]);
      float b1x = r[0] * n1, b1y = r[1] * n1, b1z = r[2] * n1;
      float d = b1x * r[3] + b1y * r[4] + b1z * r[5];
      float cx = r[3] - d * b1x, cy = r[4] - d * b1y, cz = r[5] - d * b1z;
      float n2 = rsqrtf(cx * cx + cy * cy + cz * cz);
      float b2x = cx * n2, b2y = cy * n2, b2z = cz * n2;
      float b3x = b1y * b2z - b1z * b2y, b3y = b1z * b2x - b1x * b2z, b3z = b1x * b2y - b1y * b2x;
      float* m = S.red + tid * 9;
      m[0] = b1x; m[1] = b2x; m[2] = b3x;
      m[3] = b1y; m[4] = b2y; m[5] = b3y;
      m[6] = b1z; m[7] = b2z; m[8] = b3z;
    }
    if (tid < 216) { int q = tid % 9; S.glb[tid] = (q == 0 || q == 4 || q == 8) ? 1.0f : 0.0f; }
    __syncthreads();
    if (tid < 135) S.glb[c_red[tid / 9] * 9 + tid % 9] = S.red[tid];
    __syncthreads();

    // ---- ik_R: pose[i] = glb[par]^T @ glb[i] ----
    if (tid < 24) {
      float* p = S.pose + tid * 9;
      if (tid == 0) { for (int q = 0; q < 9; ++q) p[q] = S.glb[q]; }
      else {
        const float* gp = S.glb + c_par[tid] * 9;
        const float* gi = S.glb + tid * 9;
        for (int r = 0; r < 3; ++r)
          for (int c = 0; c < 3; ++c) {
            float s = 0;
            for (int j = 0; j < 3; ++j) s += gp[j * 3 + r] * gi[j * 3 + c];
            p[r * 3 + c] = s;
          }
      }
    }
    __syncthreads();
    if (tid < 81) { int q = tid % 9; S.pose[c_ign[tid / 9] * 9 + q] = (q % 4 == 0) ? 1.0f : 0.0f; }
    __syncthreads();
    if (tid < 9) S.pose[tid] = S.RIR[tid];
    __syncthreads();

    // ---- fk_joints (serial chain, thread 0) ----
    if (tid == 0) {
      for (int q = 0; q < 9; ++q) S.Rg[q] = S.pose[q];
      S.pj[0] = S.pj[1] = S.pj[2] = 0.0f;
      for (int i = 1; i < 24; ++i) {
        int par = c_par[i];
        const float* rp = S.Rg + par * 9;
        const float* pi = S.pose + i * 9;
        float* ri = S.Rg + i * 9;
        for (int r = 0; r < 3; ++r)
          for (int c = 0; c < 3; ++c) {
            float s = 0;
            for (int j = 0; j < 3; ++j) s += rp[r * 3 + j] * pi[j * 3 + c];
            ri[r * 3 + c] = s;
          }
        for (int r = 0; r < 3; ++r) {
          float s = 0;
          for (int j = 0; j < 3; ++j) s += rp[r * 3 + j] * S.tl[i * 3 + j];
          S.pj[i * 3 + r] = S.pj[par * 3 + r] + s;
        }
      }
    }
    __syncthreads();
    if (tid < 69) {
      int k = tid / 3 + 1, i = tid % 3;
      float s = 0;
      for (int j = 0; j < 3; ++j) s += S.pj[k * 3 + j] * S.RIR[j * 3 + i];
      S.aj[tid] = s;
    }
    __syncthreads();

    // ---- imu144 -> vr1 / vr2 ----
    if (tid < 144) {
      float v;
      if (tid < 18)      v = S.aRB[tid] * 0.05f;
      else if (tid < 72) v = S.RRB[tid - 18];
      else if (tid < 75) v = S.wRB[15 + tid - 72] * 0.25f;
      else               v = S.aj[tid - 75];
      S.xv[tid] = v;
    }
    __syncthreads();
    lstm3(wsa, S, xbf, 3, S.xv, 144, 160, tid);
    headmv(wsa, S, xbf, 3, 5, tid);
    if (tid < 72) {
      int k = tid / 3, i = tid % 3;
      float s = 0;
      for (int j = 0; j < 3; ++j) s += S.yb[k * 3 + j] * S.RIR[i * 3 + j];
      S.avb[tid] = 2.0f * s;
    }
    __syncthreads();
    lstm3(wsa, S, xbf, 4, S.xv, 144, 160, tid);
    headmv(wsa, S, xbf, 4, 1, tid);
    if (tid < 2) S.cfg2[tid] = S.yb[tid];
    __syncthreads();

    // ---- outputs: pose (T,24,3,3) | av (T,24,3) | cfg (T,2) ----
    if (tid < 216) A.out[(size_t)t * 216 + tid] = S.pose[tid];
    if (tid >= 256 && tid < 328) A.out[(size_t)T_STEPS * 216 + (size_t)t * 72 + (tid - 256)] = S.avb[tid - 256];
    if (tid >= 384 && tid < 386) A.out[(size_t)T_STEPS * 288 + (size_t)t * 2 + (tid - 384)] = S.cfg2[tid - 384];

    // ---- carry update ----
    if (tid == 0) {
      for (int i = 0; i < 3; ++i) S.prevw[i] = S.wRB[15 + i];
      for (int i = 0; i < 15; ++i) { S.prevp[i] = S.pRBc[i]; S.pRBc[i] = S.pRBn[i]; }
    }
    __syncthreads();
  }
}

// ---------------------------------------------------------------------------
// Host launcher
// ---------------------------------------------------------------------------
extern "C" void kernel_launch(void* const* d_in, const int* in_sizes, int n_in,
                              void* d_out, int out_size, void* d_ws, size_t ws_size,
                              hipStream_t stream) {
  (void)in_sizes; (void)n_in; (void)out_size; (void)ws_size;
  // Input index map: JAX pytree order (sorted dict keys, lists in order).
  // 0:R 1:a 2-15:ik1 16-29:ik2 30-37:mlpA 38:pRB0 39-52:pl1 53:pl1_c0
  // 54:pl1_h0 55:t_local 56-69:vr1 70-83:vr2 84:w
  // LSTM block (base b): layer l -> Whh=b+4l, Wih=b+4l+1, bhh=b+4l+2, bih=b+4l+3; lin2 W=b+12, b=b+13
  static const int lbase[5]   = {39, 2, 16, 56, 70};   // pl1, ik1, ik2, vr1, vr2
  static const int in_act[5]  = {60, 60, 114, 144, 144};
  static const int in_pad[5]  = {64, 64, 128, 160, 160};
  static const int head_fo[5] = {15, 69, 90, 72, 2};
  static const int head_mt[5] = {1, 5, 6, 5, 1};
  static const int mlp_fo[4]  = {512, 512, 512, 15};
  static const int mlp_fi[4]  = {108, 512, 512, 512};
  static const int mlp_mt[4]  = {32, 32, 32, 1};
  static const int mlp_kt[4]  = {4, 16, 16, 16};

  constexpr WsLayout L = make_layout();
  char* ws = (char*)d_ws;

  // ---- pack LSTM layers + heads ----
  for (int s = 0; s < 5; ++s) {
    int b = lbase[s];
    for (int l = 0; l < 3; ++l) {
      int ia = (l == 0) ? in_act[s] : HDIM;
      int ip = (l == 0) ? in_pad[s] : HDIM;
      int Kp = ip + HDIM;
      size_t elems = (size_t)2048 * Kp;
      bf16* Wd = (bf16*)(ws + L.Wl[s * 3 + l]);
      float* bd = (float*)(ws + L.bl[s * 3 + l]);
      const float* Whh = (const float*)d_in[b + 4 * l + 0];
      const float* Wih = (const float*)d_in[b + 4 * l + 1];
      const float* bhh = (const float*)d_in[b + 4 * l + 2];
      const float* bih = (const float*)d_in[b + 4 * l + 3];
      pack_lstm_k<<<(int)((elems + 1023) / 1024), 1024, 0, stream>>>(Wih, Whh, ia, ip, Wd);
      bias_comb_k<<<(4 * HDIM + 255) / 256, 256, 0, stream>>>(bih, bhh, bd);
    }
    int Mt = head_mt[s];
    size_t elems = (size_t)Mt * 16 * 512;
    pack_mat_k<<<(int)((elems + 1023) / 1024), 1024, 0, stream>>>(
        (const float*)d_in[b + 12], head_fo[s], HDIM, Mt, 16, (bf16*)(ws + L.Wh[s]));
    bias_pad_k<<<1, 256, 0, stream>>>((const float*)d_in[b + 13], head_fo[s], Mt * 16,
                                      (float*)(ws + L.bh[s]));
  }

  // ---- pack mlpA ----
  for (int l = 0; l < 4; ++l) {
    size_t elems = (size_t)mlp_mt[l] * mlp_kt[l] * 512;
    pack_mat_k<<<(int)((elems + 1023) / 1024), 1024, 0, stream>>>(
        (const float*)d_in[30 + 2 * l], mlp_fo[l], mlp_fi[l], mlp_mt[l], mlp_kt[l],
        (bf16*)(ws + L.Wm[l]));
    bias_pad_k<<<(mlp_mt[l] * 16 + 255) / 256, 256, 0, stream>>>(
        (const float*)d_in[31 + 2 * l], mlp_fo[l], mlp_mt[l] * 16, (float*)(ws + L.bm[l]));
  }

  // ---- cell state ----
  cinit_k<<<(15 * HDIM + 255) / 256, 256, 0, stream>>>((const float*)d_in[53],
                                                       (float*)(ws + L.cst));

  // ---- persistent sequential scan ----
  SeqArgs A;
  A.ws = ws;
  A.R = (const float*)d_in[0];
  A.a = (const float*)d_in[1];
  A.w = (const float*)d_in[84];
  A.pRB0    = (const float*)d_in[38];
  A.pl1_h0  = (const float*)d_in[54];
  A.t_local = (const float*)d_in[55];
  A.out = (float*)d_out;
  seq_kernel<<<dim3(1), dim3(NTHR), 0, stream>>>(A);
}
